// RingAttention_65687229825781
// MI455X (gfx1250) — compile-verified, matrix operand order checked by reading, not run on hardware
//
#include <hip/hip_runtime.h>
#include <hip/hip_bf16.h>

// ---------------------------------------------------------------------------
// Causal MHA:  Q/K/V = x@W{q,k,v}.T ; O = softmax(QK^T/sqrt(dk)+mask)V ;
//              out = O@Wo.T         all matrix math on v_wmma_f32_16x16x32_bf16
// ---------------------------------------------------------------------------

typedef __bf16 bf16_t;
typedef __attribute__((ext_vector_type(16))) __bf16 v16bf;
typedef __attribute__((ext_vector_type(8)))  __bf16 v8bf;
typedef __attribute__((ext_vector_type(8)))  float  v8f;

#define B_  2
#define T_  2048
#define D_  2048
#define H_  16
#define DK_ 128

static constexpr float kScale = 0.088388347648318447f; // 1/sqrt(128)
static constexpr float kNegBig = -3.0e30f;

__device__ __forceinline__ v16bf cat16(v8bf lo, v8bf hi) {
  return __builtin_shufflevector(lo, hi, 0,1,2,3,4,5,6,7,8,9,10,11,12,13,14,15);
}

// A/B operand (16-bit, 16x32 resp. 32x16) per-lane fragment:
//   lane L: row = (L&15); elems 0..7  = K[koff .. koff+7],
//                          elems 8..15 = K[16+koff .. 23+koff], koff = (L>=16)*8
__device__ __forceinline__ v16bf load_frag(const bf16_t* __restrict__ base,
                                           int row, int ld, int k, int koff) {
  const bf16_t* p = base + (size_t)row * ld + k + koff;
  return cat16(*(const v8bf*)p, *(const v8bf*)(p + 16));
}

__device__ __forceinline__ v16bf lds_frag(const bf16_t* p) {
  return cat16(*(const v8bf*)p, *(const v8bf*)(p + 16));
}

__device__ __forceinline__ v8f wmma_bf16(v16bf a, v16bf b, v8f c) {
  return __builtin_amdgcn_wmma_f32_16x16x32_bf16(false, a, false, b, (short)0, c,
                                                 false, false);
}

// --------------------------- f32 -> bf16 conversion ------------------------
__global__ void cvt_f32_bf16(const float* __restrict__ src,
                             bf16_t* __restrict__ dst, size_t n) {
  size_t i = (size_t)blockIdx.x * blockDim.x + threadIdx.x;
  size_t stride = (size_t)gridDim.x * blockDim.x;
  for (; i < n; i += stride) dst[i] = (bf16_t)src[i];
}

// --------------------------- GEMM: Y = A * W^T -----------------------------
// A: [M][K] bf16 row-major.  W: [N][K] bf16 row-major (B[k][n] = W[n][k], so
// the B-operand column n is W row n -> contiguous per-lane loads).
// Wave tile 32x64 (2x4 WMMA accumulators), software-pipelined k-loop with
// double-buffered fragments so loads for k+32 fly while WMMAs for k execute.
// Block = 4 waves covering 64x128.
template <bool OUT_BF16>
__global__ void __launch_bounds__(128)
gemm_bf16_nt(const bf16_t* __restrict__ A, const bf16_t* __restrict__ W,
             void* __restrict__ Yv, int M, int N, int K) {
  const int lane  = threadIdx.x & 31;
  const int wave  = threadIdx.x >> 5;
  const int m0    = blockIdx.y * 64  + (wave >> 1) * 32;
  const int n0    = blockIdx.x * 128 + (wave & 1) * 64;
  const int mr    = lane & 15;
  const int koff  = (lane >> 4) << 3;
  const int rbase = (lane >> 4) << 3;

  v8f c[2][4] = {};
  v16bf a[2], b[4];
  a[0] = load_frag(A, m0 + mr,      K, 0, koff);
  a[1] = load_frag(A, m0 + 16 + mr, K, 0, koff);
  #pragma unroll
  for (int j = 0; j < 4; ++j) b[j] = load_frag(W, n0 + 16 * j + mr, K, 0, koff);

  for (int k = 0; k < K; k += 32) {
    v16bf an[2], bn[4];
    const bool more = (k + 32) < K;
    if (more) {
      an[0] = load_frag(A, m0 + mr,      K, k + 32, koff);
      an[1] = load_frag(A, m0 + 16 + mr, K, k + 32, koff);
      #pragma unroll
      for (int j = 0; j < 4; ++j)
        bn[j] = load_frag(W, n0 + 16 * j + mr, K, k + 32, koff);
    }
    #pragma unroll
    for (int i = 0; i < 2; ++i)
      #pragma unroll
      for (int j = 0; j < 4; ++j) c[i][j] = wmma_bf16(a[i], b[j], c[i][j]);
    if (more) {
      a[0] = an[0]; a[1] = an[1];
      #pragma unroll
      for (int j = 0; j < 4; ++j) b[j] = bn[j];
    }
  }

  // C/D layout: VGPR r -> row rbase+r, lane&15 -> col
  #pragma unroll
  for (int i = 0; i < 2; ++i)
    #pragma unroll
    for (int j = 0; j < 4; ++j) {
      const int col = n0 + 16 * j + mr;
      #pragma unroll
      for (int r = 0; r < 8; ++r) {
        const int row = m0 + 16 * i + rbase + r;
        if constexpr (OUT_BF16)
          ((bf16_t*)Yv)[(size_t)row * N + col] = (bf16_t)c[i][j][r];
        else
          ((float*)Yv)[(size_t)row * N + col] = c[i][j][r];
      }
    }
}

// --------------------------- Flash attention -------------------------------
// grid = (T/128, B*H); block = 256 (8 waves). Wave w owns q rows
// [qtile*128 + 16w, +16). Streams 64-key tiles with online softmax.
// K and V^T are staged cooperatively in LDS once per tile (padded rows for
// conflict-free ds_load_b128 fragment reads).
__global__ void __launch_bounds__(256)
flash_attn(const bf16_t* __restrict__ Q, const bf16_t* __restrict__ Km,
           const bf16_t* __restrict__ Vm, bf16_t* __restrict__ O) {
  __shared__ __align__(16) bf16_t Ksh[64][136];     // K tile [k][d], pad->68 dw
  __shared__ __align__(16) bf16_t Vt[DK_][72];      // V tile transposed [d][k]
  __shared__ __align__(16) bf16_t Pst[8][16][72];   // per-wave P staging

  const int lane  = threadIdx.x & 31;
  const int wave  = threadIdx.x >> 5;
  const int b     = blockIdx.y / H_;
  const int h     = blockIdx.y % H_;
  const int q0    = blockIdx.x * 128 + wave * 16;
  const int mr    = lane & 15;
  const int koff  = (lane >> 4) << 3;
  const int rbase = (lane >> 4) << 3;
  const size_t hd = (size_t)h * DK_;

  // Q fragments for this wave's 16 rows: 4 A-tiles of 16x32 over DK=128
  v16bf qa[4];
  {
    const bf16_t* qrow = Q + ((size_t)(b * T_ + q0 + mr)) * D_ + hd;
    #pragma unroll
    for (int t = 0; t < 4; ++t)
      qa[t] = cat16(*(const v8bf*)(qrow + 32 * t + koff),
                    *(const v8bf*)(qrow + 32 * t + 16 + koff));
  }

  v8f o[8] = {};
  float mi[8], li[8];
  #pragma unroll
  for (int r = 0; r < 8; ++r) { mi[r] = kNegBig; li[r] = 0.0f; }

  const int jend = blockIdx.x * 2 + 2;  // covers causal range for all 8 waves
  for (int j = 0; j < jend; ++j) {
    const int kb = j * 64;

    // ---- stage K tile (row-major) into LDS: 64x128, coalesced b128s --------
    #pragma unroll
    for (int c = 0; c < 2; ++c) {
      const int idx = c * 512 + (int)threadIdx.x * 2;  // 2 chunks per thread
      #pragma unroll
      for (int u = 0; u < 2; ++u) {
        const int kk = (idx + u) >> 4;
        const int d0 = ((idx + u) & 15) << 3;
        const bf16_t* p = Km + ((size_t)(b * T_ + kb + kk)) * D_ + hd + d0;
        *(v8bf*)&Ksh[kk][d0] = *(const v8bf*)p;
      }
    }
    // ---- stage V tile transposed into LDS (pack k-pairs into b32 stores) ---
    #pragma unroll
    for (int it = 0; it < 2; ++it) {
      const int idx = it * 256 + (int)threadIdx.x;
      const int k2  = (idx >> 4) << 1;          // even k
      const int d0  = (idx & 15) << 3;
      const bf16_t* p0 = Vm + ((size_t)(b * T_ + kb + k2)) * D_ + hd + d0;
      v8bf v0 = *(const v8bf*)p0;
      v8bf v1 = *(const v8bf*)(p0 + D_);
      #pragma unroll
      for (int e = 0; e < 8; ++e) {
        bf16_t* q = &Vt[d0 + e][k2];
        q[0] = v0[e];
        q[1] = v1[e];
      }
    }
    __syncthreads();

    // Prefetch next K/V tile rows while we compute (global_prefetch_b8)
    if (j + 1 < jend) {
      const size_t nrow = (size_t)(b * T_ + kb + 64 + (threadIdx.x & 63)) * D_ + hd;
      __builtin_prefetch(Km + nrow, 0, 1);
      __builtin_prefetch(Vm + nrow, 0, 1);
    }

    // Wave-uniform skip of fully masked key tiles (EXEC stays all-ones)
    if (kb <= q0 + 15) {
      // ---- S = (Q K^T) * scale, causal-masked ------------------------------
      v8f s[4] = {};
      #pragma unroll
      for (int sub = 0; sub < 4; ++sub) {
        const bf16_t* krow = &Ksh[16 * sub + mr][0];
        #pragma unroll
        for (int kk = 0; kk < 4; ++kk)
          s[sub] = wmma_bf16(qa[kk], lds_frag(krow + 32 * kk + koff), s[sub]);
      }
      #pragma unroll
      for (int sub = 0; sub < 4; ++sub)
        #pragma unroll
        for (int r = 0; r < 8; ++r) {
          const int qrow = q0 + rbase + r;
          const int kcol = kb + 16 * sub + mr;
          float v = s[sub][r] * kScale;
          s[sub][r] = (kcol > qrow) ? kNegBig : v;
        }

      // ---- online softmax (row stats replicated across 16-lane halves) -----
      float alpha[8];
      #pragma unroll
      for (int r = 0; r < 8; ++r) {
        float v = fmaxf(fmaxf(s[0][r], s[1][r]), fmaxf(s[2][r], s[3][r]));
        v = fmaxf(v, __shfl_xor(v, 1, 32));
        v = fmaxf(v, __shfl_xor(v, 2, 32));
        v = fmaxf(v, __shfl_xor(v, 4, 32));
        v = fmaxf(v, __shfl_xor(v, 8, 32));
        const float mnew = fmaxf(mi[r], v);
        alpha[r] = __expf(mi[r] - mnew);
        mi[r] = mnew;
      }
      #pragma unroll
      for (int sub = 0; sub < 4; ++sub)
        #pragma unroll
        for (int r = 0; r < 8; ++r)
          s[sub][r] = __expf(s[sub][r] - mi[r]);
      #pragma unroll
      for (int r = 0; r < 8; ++r) {
        float rs = s[0][r] + s[1][r] + s[2][r] + s[3][r];
        rs += __shfl_xor(rs, 1, 32);
        rs += __shfl_xor(rs, 2, 32);
        rs += __shfl_xor(rs, 4, 32);
        rs += __shfl_xor(rs, 8, 32);
        li[r] = li[r] * alpha[r] + rs;
        #pragma unroll
        for (int dt = 0; dt < 8; ++dt) o[dt][r] *= alpha[r];
      }

      // ---- P: C-layout -> LDS -> A-layout fragments ------------------------
      #pragma unroll
      for (int sub = 0; sub < 4; ++sub)
        #pragma unroll
        for (int r = 0; r < 8; ++r)
          Pst[wave][rbase + r][16 * sub + mr] = (bf16_t)s[sub][r];

      v16bf pa[2];
      #pragma unroll
      for (int pt = 0; pt < 2; ++pt)
        pa[pt] = lds_frag(&Pst[wave][mr][32 * pt + koff]);

      // ---- O += P * V  (B-operand column d = Vt row d, contiguous in LDS) --
      #pragma unroll
      for (int dt = 0; dt < 8; ++dt) {
        #pragma unroll
        for (int pt = 0; pt < 2; ++pt)
          o[dt] = wmma_bf16(pa[pt], lds_frag(&Vt[dt * 16 + mr][32 * pt + koff]),
                            o[dt]);
      }
    }
    __syncthreads();
  }

  // ---- normalize and write O (merged heads: column block h*DK) -------------
  #pragma unroll
  for (int dt = 0; dt < 8; ++dt) {
    const int col = (int)hd + dt * 16 + mr;
    #pragma unroll
    for (int r = 0; r < 8; ++r) {
      const int row = q0 + rbase + r;
      O[(size_t)(b * T_ + row) * D_ + col] = (bf16_t)(o[dt][r] / li[r]);
    }
  }
}

// --------------------------------------------------------------------------
extern "C" void kernel_launch(void* const* d_in, const int* in_sizes, int n_in,
                              void* d_out, int out_size, void* d_ws, size_t ws_size,
                              hipStream_t stream) {
  (void)in_sizes; (void)n_in; (void)out_size; (void)ws_size;
  const float* x  = (const float*)d_in[0];
  const float* Wq = (const float*)d_in[1];
  const float* Wk = (const float*)d_in[2];
  const float* Wv = (const float*)d_in[3];
  const float* Wo = (const float*)d_in[4];

  const size_t XN = (size_t)B_ * T_ * D_;   // 8.4M elems
  const size_t WN = (size_t)D_ * D_;        // 4.2M elems

  bf16_t* xb  = (bf16_t*)d_ws;
  bf16_t* wqb = xb  + XN;
  bf16_t* wkb = wqb + WN;
  bf16_t* wvb = wkb + WN;
  bf16_t* wob = wvb + WN;
  bf16_t* Qb  = wob + WN;
  bf16_t* Kb  = Qb  + XN;
  bf16_t* Vb  = Kb  + XN;
  bf16_t* Ob  = Vb  + XN;   // total ~117 MB of workspace

  // 1) convert inputs to bf16
  hipLaunchKernelGGL(cvt_f32_bf16, dim3(1024), dim3(256), 0, stream, x,  xb,  XN);
  hipLaunchKernelGGL(cvt_f32_bf16, dim3(512),  dim3(256), 0, stream, Wq, wqb, WN);
  hipLaunchKernelGGL(cvt_f32_bf16, dim3(512),  dim3(256), 0, stream, Wk, wkb, WN);
  hipLaunchKernelGGL(cvt_f32_bf16, dim3(512),  dim3(256), 0, stream, Wv, wvb, WN);
  hipLaunchKernelGGL(cvt_f32_bf16, dim3(512),  dim3(256), 0, stream, Wo, wob, WN);

  // 2) projections  (M=B*T=4096, N=K=2048)
  const int M = B_ * T_;
  dim3 gg(D_ / 128, M / 64), gb(128);
  hipLaunchKernelGGL((gemm_bf16_nt<true>),  gg, gb, 0, stream, xb, wqb, (void*)Qb, M, D_, D_);
  hipLaunchKernelGGL((gemm_bf16_nt<true>),  gg, gb, 0, stream, xb, wkb, (void*)Kb, M, D_, D_);
  hipLaunchKernelGGL((gemm_bf16_nt<true>),  gg, gb, 0, stream, xb, wvb, (void*)Vb, M, D_, D_);

  // 3) causal flash attention
  dim3 ag(T_ / 128, B_ * H_), ab(256);
  hipLaunchKernelGGL(flash_attn, ag, ab, 0, stream, Qb, Kb, Vb, Ob);

  // 4) output projection (f32 result)
  hipLaunchKernelGGL((gemm_bf16_nt<false>), gg, gb, 0, stream, Ob, wob, d_out, M, D_, D_);
}
